// CausalSelfAttention_29712583753886
// MI455X (gfx1250) — matrix operand order repaired and checked by reading, not compiled, not run on hardware
//
#include <hip/hip_runtime.h>

typedef __attribute__((ext_vector_type(16))) _Float16 v16h;
typedef __attribute__((ext_vector_type(8)))  _Float16 v8h;
typedef __attribute__((ext_vector_type(8)))  float    v8f;

#define DIMSZ 2048
#define NH    32
#define NKV   8
#define HD    64
#define SEQL  2048
#define NB    2
#define MROWS (NB * SEQL)   // 4096

static __device__ __forceinline__ v16h cat8(v8h lo, v8h hi) {
  return __builtin_shufflevector(lo, hi, 0,1,2,3,4,5,6,7,8,9,10,11,12,13,14,15);
}
static __device__ __forceinline__ v8f wmma16(v16h a, v16h b, v8f c) {
  return __builtin_amdgcn_wmma_f32_16x16x32_f16(false, a, false, b, (short)0, c, false, false);
}
static __device__ __forceinline__ float redmax16(float v) {
  v = fmaxf(v, __shfl_xor(v, 1, 32));
  v = fmaxf(v, __shfl_xor(v, 2, 32));
  v = fmaxf(v, __shfl_xor(v, 4, 32));
  v = fmaxf(v, __shfl_xor(v, 8, 32));
  return v;
}
static __device__ __forceinline__ float redsum16(float v) {
  v += __shfl_xor(v, 1, 32);
  v += __shfl_xor(v, 2, 32);
  v += __shfl_xor(v, 4, 32);
  v += __shfl_xor(v, 8, 32);
  return v;
}

// ---------------- elementwise f32 -> f16 convert ----------------
__global__ void cvt_f32_f16(const float* __restrict__ s, _Float16* __restrict__ d, int n) {
  int i = blockIdx.x * 256 + threadIdx.x;
  if (i < n) d[i] = (_Float16)s[i];
}

// ---------------- generic GEMM: C(M,N) = A(M,K) * W(N,K)^T ----------------
// A f16 row-major (M,K); W f16 row-major (N,K) == B operand natural layout.
// Block: 128 threads = 4 waves. Block tile: 256(M) x 64(N). Wave tile: 64x64
// (4 A-frags x 4 B-frags -> 16 WMMAs per K=32 step, 256B/lane operand traffic).
__global__ __launch_bounds__(128) void gemm_xwT(const _Float16* __restrict__ A,
                                                const _Float16* __restrict__ W,
                                                float* __restrict__ C,
                                                int N, int K) {
  const int lane = threadIdx.x & 31;
  const int wave = threadIdx.x >> 5;
  const int hi   = lane >> 4;      // 0/1: K-chunk selector
  const int c16  = lane & 15;
  const int n0   = blockIdx.x * 64;
  const int m0   = blockIdx.y * 256 + wave * 64;

  const _Float16* a0 = A + (size_t)(m0 + c16) * K;
  const _Float16* a1 = A + (size_t)(m0 + 16 + c16) * K;
  const _Float16* a2 = A + (size_t)(m0 + 32 + c16) * K;
  const _Float16* a3 = A + (size_t)(m0 + 48 + c16) * K;

  v8f acc[4][4] = {};
  for (int kb = 0; kb < K; kb += 32) {
    // A frags: lane holds row m; lanes0-15 get K[kb..+7],K[kb+16..+23], lanes16-31 the +8 chunks
    v16h af[4];
    af[0] = cat8(*(const v8h*)(a0 + kb + 8 * hi), *(const v8h*)(a0 + kb + 16 + 8 * hi));
    af[1] = cat8(*(const v8h*)(a1 + kb + 8 * hi), *(const v8h*)(a1 + kb + 16 + 8 * hi));
    af[2] = cat8(*(const v8h*)(a2 + kb + 8 * hi), *(const v8h*)(a2 + kb + 16 + 8 * hi));
    af[3] = cat8(*(const v8h*)(a3 + kb + 8 * hi), *(const v8h*)(a3 + kb + 16 + 8 * hi));
#pragma unroll
    for (int f = 0; f < 4; ++f) {
      // B frag: lane = column n0+16f+c16, K[kb+8*hi..+7], K[kb+16+8*hi..+7]
      const _Float16* wp = W + (size_t)(n0 + 16 * f + c16) * K + kb + 8 * hi;
      v16h bf = cat8(*(const v8h*)(wp), *(const v8h*)(wp + 16));
#pragma unroll
      for (int t = 0; t < 4; ++t) acc[t][f] = wmma16(af[t], bf, acc[t][f]);
    }
  }
#pragma unroll
  for (int t = 0; t < 4; ++t)
#pragma unroll
    for (int r = 0; r < 8; ++r) {
      float* p = C + (size_t)(m0 + 16 * t + r + 8 * hi) * N + n0 + c16;
      p[0]  = acc[t][0][r];
      p[16] = acc[t][1][r];
      p[32] = acc[t][2][r];
      p[48] = acc[t][3][r];
    }
}

// ---------------- RoPE + repack: src (B*S, nh*64) f32 -> dst (B, nh, S, 64) f16 ----------------
__global__ void rope_pack(const float* __restrict__ src, const float* __restrict__ fc,
                          const float* __restrict__ fs, _Float16* __restrict__ dst, int nh) {
  int i = blockIdx.x * 256 + threadIdx.x;        // over B*S*nh*32 rotation pairs
  int total = NB * SEQL * nh * 32;
  if (i >= total) return;
  int p = i & 31; int t = i >> 5;
  int h = t % nh; t /= nh;
  int s = t % SEQL; int b = t / SEQL;
  float c  = fc[s * 32 + p];
  float sn = fs[s * 32 + p];
  const float* q = src + (size_t)(b * SEQL + s) * (nh * HD) + h * HD + 2 * p;
  float x0 = q[0], x1 = q[1];
  _Float16* d = dst + (((size_t)(b * nh + h)) * SEQL + s) * HD + 2 * p;
  d[0] = (_Float16)(x0 * c - x1 * sn);
  d[1] = (_Float16)(x0 * sn + x1 * c);
}

// ---------------- V repack transposed: src (B*S, 512) f32 -> dst (B, KV, 64, S) f16 ----------------
__global__ void pack_vT(const float* __restrict__ src, _Float16* __restrict__ dst) {
  int i = blockIdx.x * 256 + threadIdx.x;        // over B*S*512
  if (i >= NB * SEQL * NKV * HD) return;
  int d = i & 63; int t = i >> 6;
  int kv = t & 7; t >>= 3;
  int s = t & (SEQL - 1); int b = t >> 11;
  dst[(((size_t)(b * NKV + kv)) * HD + d) * SEQL + s] =
      (_Float16)src[(size_t)(b * SEQL + s) * (NKV * HD) + kv * HD + d];
}

// ---------------- Flash attention: one wave per 16-row Q tile, key blocks of 32 ----------------
__global__ __launch_bounds__(128) void attn_kernel(const _Float16* __restrict__ qh,
                                                   const _Float16* __restrict__ kh,
                                                   const _Float16* __restrict__ vT,
                                                   _Float16* __restrict__ atth) {
  __shared__ _Float16 plds[4][16 * 32];
  const int lane = threadIdx.x & 31;
  const int wave = threadIdx.x >> 5;
  const int hi   = lane >> 4;
  const int c16  = lane & 15;

  const int tile = blockIdx.x * 4 + wave;        // 8192 tiles total
  const int qt = tile & 127;
  const int h  = (tile >> 7) & 31;
  const int b  = tile >> 12;
  const int qs = qt * 16;
  const int kv = h >> 2;

  const _Float16* qbase = qh + ((size_t)(b * NH + h)) * SEQL * HD;
  const _Float16* kbase = kh + ((size_t)(b * NKV + kv)) * SEQL * HD;
  const _Float16* vbase = vT + ((size_t)(b * NKV + kv)) * HD * SEQL;

  // Q A-frags (rows qs+c16): d-groups 0..31 and 32..63
  const _Float16* qrow = qbase + (size_t)(qs + c16) * HD;
  v16h qa0 = cat8(*(const v8h*)(qrow + 8 * hi),      *(const v8h*)(qrow + 16 + 8 * hi));
  v16h qa1 = cat8(*(const v8h*)(qrow + 32 + 8 * hi), *(const v8h*)(qrow + 48 + 8 * hi));

  v8f o[4] = {};
  float m_r[8], l_r[8];
#pragma unroll
  for (int r = 0; r < 8; ++r) { m_r[r] = -1e30f; l_r[r] = 0.0f; }

  _Float16* lp = plds[wave];
  const int kb_end = (qs + 15) >> 5;             // inclusive
  for (int kb = 0; kb <= kb_end; ++kb) {
    const int k0 = kb * 32;
    // scores: A = Q (K-dim = d), B = K^T (lane = key column, d chunks 8*hi and 16+8*hi)
    const _Float16* kp0 = kbase + (size_t)(k0 + c16) * HD + 8 * hi;
    const _Float16* kp1 = kbase + (size_t)(k0 + 16 + c16) * HD + 8 * hi;
    v16h bk00 = cat8(*(const v8h*)(kp0), *(const v8h*)(kp0 + 16));        // keys g0, d 0..31
    v16h bk01 = cat8(*(const v8h*)(kp0 + 32), *(const v8h*)(kp0 + 48));   // keys g0, d 32..63
    v16h bk10 = cat8(*(const v8h*)(kp1), *(const v8h*)(kp1 + 16));        // keys g1, d 0..31
    v16h bk11 = cat8(*(const v8h*)(kp1 + 32), *(const v8h*)(kp1 + 48));   // keys g1, d 32..63
    v8f s0 = {}, s1 = {};
    s0 = wmma16(qa0, bk00, s0); s0 = wmma16(qa1, bk01, s0);
    s1 = wmma16(qa0, bk10, s1); s1 = wmma16(qa1, bk11, s1);

    const bool diag = (k0 + 31 > qs);            // wave-uniform
#pragma unroll
    for (int r = 0; r < 8; ++r) {
      float a  = s0[r] * 0.125f;                 // 1/sqrt(64)
      float bb = s1[r] * 0.125f;
      if (diag) {
        int row = qs + r + 8 * hi;
        a  = (k0 + c16      > row) ? -1e30f : a;
        bb = (k0 + 16 + c16 > row) ? -1e30f : bb;
      }
      float mx   = redmax16(fmaxf(a, bb));
      float mnew = fmaxf(m_r[r], mx);
      float alpha = __expf(m_r[r] - mnew);
      m_r[r] = mnew;
      float p0 = __expf(a - mnew);
      float p1 = __expf(bb - mnew);
      s0[r] = p0; s1[r] = p1;
      l_r[r] = l_r[r] * alpha + redsum16(p0 + p1);
      o[0][r] *= alpha; o[1][r] *= alpha; o[2][r] *= alpha; o[3][r] *= alpha;
    }

    // C-layout -> A-layout for P via per-wave LDS staging
#pragma unroll
    for (int r = 0; r < 8; ++r) {
      int row = r + 8 * hi;
      lp[row * 32 + c16]      = (_Float16)s0[r];
      lp[row * 32 + 16 + c16] = (_Float16)s1[r];
    }
    asm volatile("s_wait_dscnt 0" ::: "memory");
    v16h pa = cat8(*(const v8h*)(lp + c16 * 32 + 8 * hi),
                   *(const v8h*)(lp + c16 * 32 + 16 + 8 * hi));

    // O += P(16x32) * V(32x64): B frag lane = dv column, keys k0+8*hi..+7 and k0+16+8*hi..+7
#pragma unroll
    for (int f = 0; f < 4; ++f) {
      const _Float16* vp = vbase + (size_t)(16 * f + c16) * SEQL + k0 + 8 * hi;
      v16h vb = cat8(*(const v8h*)(vp), *(const v8h*)(vp + 16));
      o[f] = wmma16(pa, vb, o[f]);
    }
  }

#pragma unroll
  for (int r = 0; r < 8; ++r) {
    float inv = 1.0f / l_r[r];
    int row = qs + r + 8 * hi;
    _Float16* op = atth + ((size_t)(b * SEQL + row)) * (NH * HD) + h * HD + c16;
    op[0]  = (_Float16)(o[0][r] * inv);
    op[16] = (_Float16)(o[1][r] * inv);
    op[32] = (_Float16)(o[2][r] * inv);
    op[48] = (_Float16)(o[3][r] * inv);
  }
}

extern "C" void kernel_launch(void* const* d_in, const int* in_sizes, int n_in,
                              void* d_out, int out_size, void* d_ws, size_t ws_size,
                              hipStream_t stream) {
  const float* x  = (const float*)d_in[0];
  const float* fc = (const float*)d_in[1];
  const float* fs = (const float*)d_in[2];
  const float* wq = (const float*)d_in[3];
  const float* wk = (const float*)d_in[4];
  const float* wv = (const float*)d_in[5];
  const float* wo = (const float*)d_in[6];
  float* out = (float*)d_out;

  char* ws = (char*)d_ws;
  size_t off = 0;
  auto give = [&](size_t bytes) -> char* {
    char* p = ws + off;
    off = (off + bytes + 255) & ~(size_t)255;
    return p;
  };

  const int nX  = MROWS * DIMSZ;            // 8388608
  const int nWq = DIMSZ * DIMSZ;            // 4194304
  const int nWk = NKV * HD * DIMSZ;         // 1048576

  _Float16* xh  = (_Float16*)give((size_t)nX * 2);
  _Float16* wqh = (_Float16*)give((size_t)nWq * 2);
  _Float16* wkh = (_Float16*)give((size_t)nWk * 2);
  _Float16* wvh = (_Float16*)give((size_t)nWk * 2);
  _Float16* woh = (_Float16*)give((size_t)nWq * 2);
  float* q32 = (float*)give((size_t)MROWS * DIMSZ * 4);
  float* k32 = (float*)give((size_t)MROWS * (NKV * HD) * 4);
  float* v32 = (float*)give((size_t)MROWS * (NKV * HD) * 4);
  _Float16* qh = (_Float16*)give((size_t)NB * NH * SEQL * HD * 2);
  _Float16* kh = (_Float16*)give((size_t)NB * NKV * SEQL * HD * 2);
  _Float16* vT = (_Float16*)give((size_t)NB * NKV * SEQL * HD * 2);
  _Float16* atth = (_Float16*)q32;          // reuse: q32 dead after rope_pack

  // 1) fp32 -> f16 conversions
  cvt_f32_f16<<<nX / 256, 256, 0, stream>>>(x,  xh,  nX);
  cvt_f32_f16<<<nWq / 256, 256, 0, stream>>>(wq, wqh, nWq);
  cvt_f32_f16<<<nWk / 256, 256, 0, stream>>>(wk, wkh, nWk);
  cvt_f32_f16<<<nWk / 256, 256, 0, stream>>>(wv, wvh, nWk);
  cvt_f32_f16<<<nWq / 256, 256, 0, stream>>>(wo, woh, nWq);

  // 2) QKV projections (WMMA GEMMs)
  gemm_xwT<<<dim3(DIMSZ / 64, MROWS / 256), 128, 0, stream>>>(xh, wqh, q32, DIMSZ, DIMSZ);
  gemm_xwT<<<dim3((NKV * HD) / 64, MROWS / 256), 128, 0, stream>>>(xh, wkh, k32, NKV * HD, DIMSZ);
  gemm_xwT<<<dim3((NKV * HD) / 64, MROWS / 256), 128, 0, stream>>>(xh, wvh, v32, NKV * HD, DIMSZ);

  // 3) RoPE + repack to attention layouts
  rope_pack<<<(NB * SEQL * NH * 32) / 256, 256, 0, stream>>>(q32, fc, fs, qh, NH);
  rope_pack<<<(NB * SEQL * NKV * 32) / 256, 256, 0, stream>>>(k32, fc, fs, kh, NKV);
  pack_vT<<<(NB * SEQL * NKV * HD) / 256, 256, 0, stream>>>(v32, vT);

  // 4) causal flash attention (WMMA)
  attn_kernel<<<(NB * NH * (SEQL / 16)) / 4, 128, 0, stream>>>(qh, kh, vT, atth);

  // 5) output projection straight into d_out (f32)
  gemm_xwT<<<dim3(DIMSZ / 64, MROWS / 256), 128, 0, stream>>>(atth, woh, out, DIMSZ, DIMSZ);
}
